// IDRModel_33191507264052
// MI455X (gfx1250) — matrix operand order repaired and checked by reading, not compiled, not run on hardware
//
#include <hip/hip_runtime.h>
#include <hip/hip_bf16.h>
#include <math.h>

typedef __attribute__((ext_vector_type(16))) __bf16 v16bf;
typedef __attribute__((ext_vector_type(8)))  __bf16 v8bf;
typedef __attribute__((ext_vector_type(4)))  __bf16 v4bf;
typedef __attribute__((ext_vector_type(8)))  float  v8f;
typedef __attribute__((ext_vector_type(4)))  float  f32x4;

__constant__ int kScales[7] = {1, 3, 4, 5, 8, 16, 32};

// ---------------------------------------------------------------------------
// Generic GEMM: C(M,N) = op( A(M,K) @ W(K,N) + bias[N] )  via bf16 WMMA
// op: 0 = none, 1 = sigmoid, 2 = add residual res(M,N)
// Block tile 128(M) x 128(N), K step 32. 256 threads = 8 waves (4x2 grid),
// each wave owns a 32x64 macro-tile = 8 v_wmma per K-step.
// Double-buffered LDS; B tile stored pre-swizzled in WMMA fragment order so
// every fragment load is a contiguous ds_load_b128 pair.
// ---------------------------------------------------------------------------
#define BM 128
#define BN 128
#define KT 32

__global__ __launch_bounds__(256)
void gemm_bf16_wmma(const float* __restrict__ A, const float* __restrict__ W,
                    const float* __restrict__ bias, const float* __restrict__ res,
                    float* __restrict__ C, int M, int N, int K, int op)
{
    // sA[st][row][k]            : A rows contiguous (frag = 2x ds_load_b128)
    // sB[st][colgrp][lane][16]  : fragment-order swizzle (frag = 2x ds_load_b128)
    __shared__ __align__(16) __bf16 sA[2][BM][KT];       // 16 KB
    __shared__ __align__(16) __bf16 sB[2][8][32][16];    // 16 KB

    const int tid  = threadIdx.x;
    const int wave = tid >> 5;
    const int lane = tid & 31;
    const int wm   = wave >> 1;          // 0..3  (row of 32)
    const int wn   = wave & 1;           // 0..1  (col of 64)
    const int bm0  = blockIdx.y * BM;
    const int bn0  = blockIdx.x * BN;
    const int mlan = lane & 15;
    const int hi   = lane >> 4;

    // staging thread mapping
    const int ar  = tid >> 3;            // A row 0..31 (+32/+64/+96)
    const int ac  = (tid & 7) * 4;       // A col group of 4
    const int bg  = tid >> 5;            // B col group 0..7
    const int bl  = tid & 31;            // B dest lane
    const int bn_ = bl & 15;
    const int bhi = bl >> 4;

    auto stageA = [&](int st, int k0) {
        #pragma unroll
        for (int rr = 0; rr < 4; ++rr) {
            f32x4 a = *(const f32x4*)&A[(size_t)(bm0 + ar + rr * 32) * K + k0 + ac];
            v4bf p;
            #pragma unroll
            for (int j = 0; j < 4; ++j) p[j] = (__bf16)a[j];
            *(v4bf*)&sA[st][ar + rr * 32][ac] = p;
        }
    };
    auto stageB = [&](int st, int k0) {
        const float* wp = &W[(size_t)(k0 + bhi * 8) * N + bn0 + bg * 16 + bn_];
        v8bf t0, t1;
        #pragma unroll
        for (int j = 0; j < 8; ++j) t0[j] = (__bf16)wp[(size_t)j * N];
        #pragma unroll
        for (int j = 0; j < 8; ++j) t1[j] = (__bf16)wp[(size_t)(16 + j) * N];
        *(v8bf*)&sB[st][bg][bl][0] = t0;
        *(v8bf*)&sB[st][bg][bl][8] = t1;
    };

    v8f acc[2][4];
    #pragma unroll
    for (int i = 0; i < 2; ++i)
        #pragma unroll
        for (int j = 0; j < 4; ++j) acc[i][j] = (v8f)0.0f;

    stageA(0, 0);
    stageB(0, 0);
    __syncthreads();

    int st = 0;
    for (int k0 = 0; k0 < K; k0 += KT, st ^= 1) {
        if (k0 + KT < K) {               // stage next tile into the other buffer
            stageA(st ^ 1, k0 + KT);
            stageB(st ^ 1, k0 + KT);
        }
        if (k0 + 2 * KT < K) {           // pull tile-after-next toward the WGP
            __builtin_prefetch(&A[(size_t)(bm0 + ar) * K + k0 + 2 * KT + ac], 0, 1);
            __builtin_prefetch(&W[(size_t)(k0 + 2 * KT + (tid >> 4)) * N + bn0 + (tid & 15) * 8], 0, 1);
        }

        v16bf af[2], bv[4];
        #pragma unroll
        for (int im = 0; im < 2; ++im) {
            int row = wm * 32 + im * 16 + mlan;
            v8bf lo = *(const v8bf*)&sA[st][row][hi * 8];
            v8bf hh = *(const v8bf*)&sA[st][row][16 + hi * 8];
            af[im] = __builtin_shufflevector(lo, hh, 0, 1, 2, 3, 4, 5, 6, 7,
                                             8, 9, 10, 11, 12, 13, 14, 15);
        }
        #pragma unroll
        for (int in = 0; in < 4; ++in)
            bv[in] = *(const v16bf*)&sB[st][4 * wn + in][lane][0];

        #pragma unroll
        for (int im = 0; im < 2; ++im)
            #pragma unroll
            for (int in = 0; in < 4; ++in)
                acc[im][in] = __builtin_amdgcn_wmma_f32_16x16x32_bf16(
                    false, af[im], false, bv[in], (short)0, acc[im][in], false, false);
        __syncthreads();
    }

    // epilogue: bias (+ sigmoid | + residual), store f32
    #pragma unroll
    for (int im = 0; im < 2; ++im)
        #pragma unroll
        for (int in = 0; in < 4; ++in) {
            int n = bn0 + wn * 64 + in * 16 + mlan;
            float bvv = bias[n];
            #pragma unroll
            for (int v = 0; v < 8; ++v) {
                int m = bm0 + wm * 32 + im * 16 + v + 8 * hi;
                float val = acc[im][in][v] + bvv;
                if (op == 1) val = 1.0f / (1.0f + expf(-val));
                else if (op == 2) val += res[(size_t)m * N + n];
                C[(size_t)m * N + n] = val;
            }
        }
}

// ---------------------------------------------------------------------------
// Complex CWT: dilated 5-tap conv over L, per scale.  Output layout (B,L,S,128)
// ---------------------------------------------------------------------------
__global__ __launch_bounds__(128)
void cwt_kernel(const float* __restrict__ x,   // (B, L, 128)
                const float* __restrict__ kr,  // (S, 128, 128, 5)
                const float* __restrict__ ki,
                float* __restrict__ Wr, float* __restrict__ Wi, int L)
{
    __shared__ float sx[5][128];
    const int l = blockIdx.x, s = blockIdx.y, b = blockIdx.z;
    const int co = threadIdx.x;
    const int sc = kScales[s];
    #pragma unroll
    for (int k = 0; k < 5; ++k) {
        int lp = l + (k - 2) * sc;
        sx[k][co] = (lp >= 0 && lp < L) ? x[((size_t)b * L + lp) * 128 + co] : 0.0f;
    }
    __syncthreads();
    float ar = 0.f, ai = 0.f;
    const float* krp = kr + ((size_t)s * 128 + co) * 128 * 5;
    const float* kip = ki + ((size_t)s * 128 + co) * 128 * 5;
    for (int ci = 0; ci < 128; ++ci) {
        #pragma unroll
        for (int k = 0; k < 5; ++k) {
            float xv = sx[k][ci];
            ar = fmaf(xv, krp[ci * 5 + k], ar);
            ai = fmaf(xv, kip[ci * 5 + k], ai);
        }
    }
    size_t o = (((size_t)b * L + l) * 7 + s) * 128 + co;
    Wr[o] = ar; Wi[o] = ai;
}

// ---------------------------------------------------------------------------
// Per-token scale attention: K/V proj (Dw=128 -> Da=128) over S=7 scales,
// softmax over scales, context.
// ---------------------------------------------------------------------------
__global__ __launch_bounds__(128)
void scale_att_kernel(const float* __restrict__ Q,   // (T, 128)
                      const float* __restrict__ Wp,  // (T, 7, 128)
                      const float* __restrict__ wk, const float* __restrict__ bk,
                      const float* __restrict__ wv, const float* __restrict__ bv,
                      float* __restrict__ ctx, int T)
{
    __shared__ float sQ[128];
    __shared__ float sW[7][128];
    __shared__ float sRed[7][128];
    __shared__ float sAtt[7];
    const int t = blockIdx.x, a = threadIdx.x;
    sQ[a] = Q[(size_t)t * 128 + a];
    #pragma unroll
    for (int s = 0; s < 7; ++s) sW[s][a] = Wp[((size_t)t * 7 + s) * 128 + a];
    __syncthreads();
    float Ks[7], Vs[7];
    #pragma unroll
    for (int s = 0; s < 7; ++s) { Ks[s] = bk[a]; Vs[s] = bv[a]; }
    for (int d = 0; d < 128; ++d) {
        float wkv = wk[d * 128 + a], wvv = wv[d * 128 + a];
        #pragma unroll
        for (int s = 0; s < 7; ++s) {
            Ks[s] = fmaf(sW[s][d], wkv, Ks[s]);
            Vs[s] = fmaf(sW[s][d], wvv, Vs[s]);
        }
    }
    #pragma unroll
    for (int s = 0; s < 7; ++s) sRed[s][a] = sQ[a] * Ks[s];
    __syncthreads();
    if (a < 7) {
        float e = 0.f;
        for (int j = 0; j < 128; ++j) e += sRed[a][j];
        sAtt[a] = e * 0.08838834764831845f;  // 1/sqrt(128)
    }
    __syncthreads();
    if (a == 0) {
        float mx = sAtt[0];
        #pragma unroll
        for (int s = 1; s < 7; ++s) mx = fmaxf(mx, sAtt[s]);
        float sum = 0.f;
        #pragma unroll
        for (int s = 0; s < 7; ++s) { sAtt[s] = expf(sAtt[s] - mx); sum += sAtt[s]; }
        float inv = 1.0f / sum;
        #pragma unroll
        for (int s = 0; s < 7; ++s) sAtt[s] *= inv;
    }
    __syncthreads();
    float c = 0.f;
    #pragma unroll
    for (int s = 0; s < 7; ++s) c = fmaf(sAtt[s], Vs[s], c);
    ctx[(size_t)t * 128 + a] = c;
}

// ---------------------------------------------------------------------------
// Complex interaction + concat -> cat (T, 256)
// ---------------------------------------------------------------------------
__global__ void cib_kernel(const float* __restrict__ ctxR, const float* __restrict__ ctxI,
                           const float* __restrict__ ga, const float* __restrict__ gb,
                           float* __restrict__ cat, int T)
{
    int i = blockIdx.x * blockDim.x + threadIdx.x;
    if (i >= T * 128) return;
    int t = i >> 7, a = i & 127;
    float r = ctxR[i], im = ctxI[i];
    cat[(size_t)t * 256 + a]       = ga[a] * r - gb[a] * im;
    cat[(size_t)t * 256 + 128 + a] = gb[a] * r + ga[a] * im;
}

// ---------------------------------------------------------------------------
// dt: per-channel mean(softplus) over 4096 rows, clamped [1e-3, 2]
// ---------------------------------------------------------------------------
__global__ __launch_bounds__(256)
void dt_reduce_kernel(const float* __restrict__ dtlin, float* __restrict__ dt,
                      int rows, int H)
{
    __shared__ float red[256];
    const int c = blockIdx.x;
    float s = 0.f;
    for (int r = threadIdx.x; r < rows; r += 256) {
        float v = dtlin[(size_t)r * H + c];
        s += (v > 20.f) ? v : log1pf(expf(v));
    }
    red[threadIdx.x] = s; __syncthreads();
    for (int st = 128; st > 0; st >>= 1) {
        if (threadIdx.x < st) red[threadIdx.x] += red[threadIdx.x + st];
        __syncthreads();
    }
    if (threadIdx.x == 0)
        dt[c] = fminf(fmaxf(red[0] / (float)rows, 1e-3f), 2.0f);
}

__global__ void adbd_kernel(const float* __restrict__ A_raw, const float* __restrict__ Bm,
                            const float* __restrict__ dt,
                            float* __restrict__ Ad, float* __restrict__ Bd, int n)
{
    int i = blockIdx.x * blockDim.x + threadIdx.x;
    if (i >= n) return;
    int c = i >> 4;
    float ar = A_raw[i];
    float sp = (ar > 20.f) ? ar : log1pf(expf(ar));
    float A = -sp;
    float ad = expf(A * dt[c]);
    Ad[i] = ad;
    Bd[i] = (ad - 1.0f) / (A + 1e-6f) * Bm[i];
}

// depthwise conv(k=3) on u + gate mix: u2 = u*g + h*(1-g)
__global__ void convgate_kernel(const float* __restrict__ u, const float* __restrict__ g,
                                const float* __restrict__ dww, const float* __restrict__ dwb,
                                float* __restrict__ u2, int B_, int L_, int H_)
{
    size_t i = (size_t)blockIdx.x * blockDim.x + threadIdx.x;
    if (i >= (size_t)B_ * L_ * H_) return;
    int c = (int)(i % H_);
    int l = (int)((i / H_) % L_);
    float h = dwb[c];
    #pragma unroll
    for (int k = 0; k < 3; ++k) {
        int lp = l + k - 1;
        if (lp >= 0 && lp < L_)
            h = fmaf(u[i + (size_t)(k - 1) * H_], dww[c * 3 + k], h);
    }
    float uu = u[i], gg = g[i];
    u2[i] = uu * gg + h * (1.0f - gg);
}

// selective scan: A_d/B_d are time-invariant -> 8192 independent recurrences
__global__ __launch_bounds__(256)
void scan_kernel(const float* __restrict__ u2, const float* __restrict__ Ad,
                 const float* __restrict__ Bd, const float* __restrict__ Cm,
                 float* __restrict__ y, int B_, int L_, int H_)
{
    int gid = blockIdx.x * blockDim.x + threadIdx.x;
    if (gid >= B_ * H_) return;
    int b = gid / H_, c = gid % H_;
    float ad[16], bd[16], cc[16], h[16];
    #pragma unroll
    for (int s = 0; s < 16; ++s) {
        ad[s] = Ad[c * 16 + s]; bd[s] = Bd[c * 16 + s];
        cc[s] = Cm[c * 16 + s]; h[s] = 0.f;
    }
    const float* up = u2 + (size_t)b * L_ * H_ + c;
    float* yp = y + (size_t)b * L_ * H_ + c;
    for (int t = 0; t < L_; ++t) {
        float uv = up[(size_t)t * H_];
        float acc = 0.f;
        #pragma unroll
        for (int s = 0; s < 16; ++s) {
            h[s] = fmaf(ad[s], h[s], bd[s] * uv);
            acc = fmaf(h[s], cc[s], acc);
        }
        yp[(size_t)t * H_] = acc;
    }
}

// layernorm(H + delta) -> H (in place, one token per block, D=1024)
__global__ __launch_bounds__(256)
void ln_residual_kernel(float* __restrict__ Hb, const float* __restrict__ delta,
                        const float* __restrict__ gam, const float* __restrict__ bet)
{
    __shared__ float red[256];
    const int t = blockIdx.x, tid = threadIdx.x;
    float x[4];
    float s = 0.f;
    #pragma unroll
    for (int j = 0; j < 4; ++j) {
        int d = tid + j * 256;
        x[j] = Hb[(size_t)t * 1024 + d] + delta[(size_t)t * 1024 + d];
        s += x[j];
    }
    red[tid] = s; __syncthreads();
    for (int st = 128; st > 0; st >>= 1) {
        if (tid < st) red[tid] += red[tid + st];
        __syncthreads();
    }
    float mu = red[0] * (1.0f / 1024.0f);
    __syncthreads();
    float sq = 0.f;
    #pragma unroll
    for (int j = 0; j < 4; ++j) { float dd = x[j] - mu; sq += dd * dd; }
    red[tid] = sq; __syncthreads();
    for (int st = 128; st > 0; st >>= 1) {
        if (tid < st) red[tid] += red[tid + st];
        __syncthreads();
    }
    float inv = rsqrtf(red[0] * (1.0f / 1024.0f) + 1e-5f);
    #pragma unroll
    for (int j = 0; j < 4; ++j) {
        int d = tid + j * 256;
        Hb[(size_t)t * 1024 + d] = (x[j] - mu) * inv * gam[d] + bet[d];
    }
}

// 4 head dot-products per token -> out (4, T)
__global__ __launch_bounds__(256)
void heads_kernel(const float* __restrict__ Hb,
                  const float* w0, const float* b0, const float* w1, const float* b1,
                  const float* w2, const float* b2, const float* w3, const float* b3,
                  float* __restrict__ out, int T)
{
    __shared__ float red[4][256];
    const int t = blockIdx.x, tid = threadIdx.x;
    float p0 = 0, p1 = 0, p2 = 0, p3 = 0;
    for (int d = tid; d < 1024; d += 256) {
        float h = Hb[(size_t)t * 1024 + d];
        p0 = fmaf(h, w0[d], p0); p1 = fmaf(h, w1[d], p1);
        p2 = fmaf(h, w2[d], p2); p3 = fmaf(h, w3[d], p3);
    }
    red[0][tid] = p0; red[1][tid] = p1; red[2][tid] = p2; red[3][tid] = p3;
    __syncthreads();
    for (int st = 128; st > 0; st >>= 1) {
        if (tid < st) {
            #pragma unroll
            for (int k = 0; k < 4; ++k) red[k][tid] += red[k][tid + st];
        }
        __syncthreads();
    }
    if (tid == 0) {
        out[0 * T + t] = red[0][0] + b0[0];
        out[1 * T + t] = red[1][0] + b1[0];
        out[2 * T + t] = red[2][0] + b2[0];
        out[3 * T + t] = red[3][0] + b3[0];
    }
}

// ---------------------------------------------------------------------------
extern "C" void kernel_launch(void* const* d_in, const int* in_sizes, int n_in,
                              void* d_out, int out_size, void* d_ws, size_t ws_size,
                              hipStream_t stream)
{
    (void)in_sizes; (void)n_in; (void)out_size; (void)ws_size;
    const int T = 4096, D = 1024, HH = 2048, Bb = 4, Ll = 1024;

    const float* emb  = (const float*)d_in[0];
    const float* drw  = (const float*)d_in[1];
    const float* drb  = (const float*)d_in[2];
    const float* kerr = (const float*)d_in[3];
    const float* keri = (const float*)d_in[4];
    // real_att: wq,bq,wk,bk,wv,bv at 5..10 ; imag_att at 11..16
    const float* r_wq = (const float*)d_in[5];  const float* r_bq = (const float*)d_in[6];
    const float* r_wk = (const float*)d_in[7];  const float* r_bk = (const float*)d_in[8];
    const float* r_wv = (const float*)d_in[9];  const float* r_bv = (const float*)d_in[10];
    const float* i_wq = (const float*)d_in[11]; const float* i_bq = (const float*)d_in[12];
    const float* i_wk = (const float*)d_in[13]; const float* i_bk = (const float*)d_in[14];
    const float* i_wv = (const float*)d_in[15]; const float* i_bv = (const float*)d_in[16];
    const float* ciba = (const float*)d_in[17];
    const float* cibb = (const float*)d_in[18];
    const float* fw   = (const float*)d_in[19];
    const float* fb   = (const float*)d_in[20];
    const float* opw  = (const float*)d_in[21];
    const float* opb  = (const float*)d_in[22];
    // heads 23..30 (w,b x4: disorder, protein, rna, dna); mamba base 31, 15 each

    float* ws = (float*)d_ws;
    const size_t M8 = (size_t)8u << 20;   // 8M floats
    const size_t M4 = (size_t)4u << 20;   // 4M floats
    const size_t HK = (size_t)1u << 19;   // 512K floats
    float* buf0 = ws;                     // u / early-phase small buffers
    float* buf1 = ws + M8;                // Wr / g / y(scan out)
    float* buf2 = ws + 2 * M8;            // Wi / dtlin / u2
    float* buf3 = ws + 3 * M8;            // mamba output delta (4M)
    float* Hbuf = ws + 3 * M8 + M4;       // residual stream (4M)
    float* dt   = ws + 3 * M8 + 2 * M4;   // 2048
    float* Ad   = dt + 2048;              // 32768
    float* Bd   = Ad + 32768;             // 32768

    float* xbuf = buf0;
    float* Qr   = buf0 + HK;
    float* Qi   = buf0 + 2 * HK;
    float* ctxR = buf0 + 3 * HK;
    float* ctxI = buf0 + 4 * HK;
    float* cat  = buf0 + 5 * HK;          // 1M floats
    float* fuse = buf0 + 7 * HK;

    dim3 blk(256);

    // 1) dim reduce: x = emb @ drw + drb    (4096x1024 @ 1024x128)
    gemm_bf16_wmma<<<dim3(1, T / BM), blk, 0, stream>>>(emb, drw, drb, nullptr, xbuf, T, 128, D, 0);

    // 2) complex CWT -> Wr, Wi in (B,L,S,128)
    cwt_kernel<<<dim3(Ll, 7, Bb), dim3(128), 0, stream>>>(xbuf, kerr, keri, buf1, buf2, Ll);

    // 3) Q projections
    gemm_bf16_wmma<<<dim3(1, T / BM), blk, 0, stream>>>(emb, r_wq, r_bq, nullptr, Qr, T, 128, D, 0);
    gemm_bf16_wmma<<<dim3(1, T / BM), blk, 0, stream>>>(emb, i_wq, i_bq, nullptr, Qi, T, 128, D, 0);

    // 4) scale attention (real, imag)
    scale_att_kernel<<<dim3(T), dim3(128), 0, stream>>>(Qr, buf1, r_wk, r_bk, r_wv, r_bv, ctxR, T);
    scale_att_kernel<<<dim3(T), dim3(128), 0, stream>>>(Qi, buf2, i_wk, i_bk, i_wv, i_bv, ctxI, T);

    // 5) complex interaction + concat
    cib_kernel<<<dim3((T * 128 + 255) / 256), blk, 0, stream>>>(ctxR, ctxI, ciba, cibb, cat, T);

    // 6) fuse: (T,256)@(256,128)
    gemm_bf16_wmma<<<dim3(1, T / BM), blk, 0, stream>>>(cat, fw, fb, nullptr, fuse, T, 128, 256, 0);

    // 7) H = emb + fuse @ out_proj  (residual epilogue)
    gemm_bf16_wmma<<<dim3(D / BN, T / BM), blk, 0, stream>>>(fuse, opw, opb, emb, Hbuf, T, D, 128, 2);

    // 8) 3 x Mamba
    for (int layer = 0; layer < 3; ++layer) {
        int base = 31 + layer * 15;
        const float* x_w = (const float*)d_in[base + 0];
        const float* x_b = (const float*)d_in[base + 1];
        const float* g_w = (const float*)d_in[base + 2];
        const float* g_b = (const float*)d_in[base + 3];
        const float* dww = (const float*)d_in[base + 4];
        const float* dwb = (const float*)d_in[base + 5];
        const float* Araw= (const float*)d_in[base + 6];
        const float* Bm  = (const float*)d_in[base + 7];
        const float* Cm  = (const float*)d_in[base + 8];
        const float* dtw = (const float*)d_in[base + 9];
        const float* dtb = (const float*)d_in[base + 10];
        const float* o_w = (const float*)d_in[base + 11];
        const float* o_b = (const float*)d_in[base + 12];
        const float* lng = (const float*)d_in[base + 13];
        const float* lnb = (const float*)d_in[base + 14];

        // u, g, dtlin: three (4096x1024)@(1024x2048) GEMMs
        gemm_bf16_wmma<<<dim3(HH / BN, T / BM), blk, 0, stream>>>(Hbuf, x_w, x_b, nullptr, buf0, T, HH, D, 0);
        gemm_bf16_wmma<<<dim3(HH / BN, T / BM), blk, 0, stream>>>(Hbuf, g_w, g_b, nullptr, buf1, T, HH, D, 1);
        gemm_bf16_wmma<<<dim3(HH / BN, T / BM), blk, 0, stream>>>(Hbuf, dtw, dtb, nullptr, buf2, T, HH, D, 0);

        dt_reduce_kernel<<<dim3(HH), blk, 0, stream>>>(buf2, dt, T, HH);
        adbd_kernel<<<dim3((HH * 16 + 255) / 256), blk, 0, stream>>>(Araw, Bm, dt, Ad, Bd, HH * 16);

        // depthwise conv + gate -> u2 (overwrites dtlin, already consumed)
        convgate_kernel<<<dim3((unsigned)(((size_t)Bb * Ll * HH + 255) / 256)), blk, 0, stream>>>(
            buf0, buf1, dww, dwb, buf2, Bb, Ll, HH);

        // scan -> y (overwrites g, already consumed)
        scan_kernel<<<dim3((Bb * HH + 255) / 256), blk, 0, stream>>>(buf2, Ad, Bd, Cm, buf1, Bb, Ll, HH);

        // out projection (4096x2048)@(2048x1024)
        gemm_bf16_wmma<<<dim3(D / BN, T / BM), blk, 0, stream>>>(buf1, o_w, o_b, nullptr, buf3, T, D, HH, 0);

        // H = layernorm(H + delta)
        ln_residual_kernel<<<dim3(T), blk, 0, stream>>>(Hbuf, buf3, lng, lnb);
    }

    // 9) heads -> (4, B, L)
    heads_kernel<<<dim3(T), blk, 0, stream>>>(Hbuf,
        (const float*)d_in[23], (const float*)d_in[24],
        (const float*)d_in[25], (const float*)d_in[26],
        (const float*)d_in[27], (const float*)d_in[28],
        (const float*)d_in[29], (const float*)d_in[30],
        (float*)d_out, T);
}
